// ChongTayConvLstmBottleneckBlock_48009144435007
// MI455X (gfx1250) — matrix operand order repaired and checked by reading, not compiled, not run on hardware
//
#include <hip/hip_runtime.h>

typedef __attribute__((ext_vector_type(16))) _Float16 v16h;
typedef __attribute__((ext_vector_type(8)))  _Float16 v8h;
typedef __attribute__((ext_vector_type(8)))  float    v8f;

static constexpr int Bn = 4, Tn = 16, Hn = 64, Wn = 64;

__device__ __forceinline__ float hsig(float x) {
    return fminf(fmaxf(0.2f * x + 0.5f, 0.0f), 1.0f);
}

// A-fragment load from an LDS halo patch (16-bit 16x32 layout):
// lanes 0-15 take K=0..7 & 16..23, lanes 16-31 take K=8..15 & 24..31.
__device__ __forceinline__ v16h load_a(const _Float16* s, int row, int col,
                                       int cstride, int cb) {
    int base = (row * 18 + col) * cstride + cb;
    v8h lo = *(const v8h*)(s + base);
    v8h hi = *(const v8h*)(s + base + 16);
    return __builtin_shufflevector(lo, hi, 0, 1, 2, 3, 4, 5, 6, 7,
                                   8, 9, 10, 11, 12, 13, 14, 15);
}

// ---------------------------------------------------------------- pre-passes

__global__ __launch_bounds__(256)
void cvt_f32_to_f16(const float* __restrict__ src, _Float16* __restrict__ dst, int n) {
    int i = blockIdx.x * 256 + threadIdx.x;
    if (i < n) dst[i] = (_Float16)src[i];
}

// w: [3][3][K=CIN][N=FOUT4] f32  ->  pw: [9][CIN/32][FOUT4/16][32 lanes][16 halves]
// B-fragment layout (16-bit, 32x16): lanes 0-15 hold K=0..7 & 16..23, lanes 16-31 hold K=8..15 & 24..31.
template<int CIN, int FOUT4>
__global__ __launch_bounds__(256)
void pack_w(const float* __restrict__ w, _Float16* __restrict__ pw) {
    int idx = blockIdx.x * 256 + threadIdx.x;
    if (idx >= 9 * CIN * FOUT4) return;
    constexpr int NT = FOUT4 / 16;
    constexpr int KC = CIN / 32;
    int h    = idx & 15;
    int r    = idx >> 4;
    int lane = r & 31; r >>= 5;
    int nt   = r % NT; r /= NT;
    int kc   = r % KC;
    int tap  = r / KC;
    int N = nt * 16 + (lane & 15);
    int K = kc * 32 + ((h & 8) << 1) + ((lane >> 4) << 3) + (h & 7);
    pw[idx] = (_Float16)w[((size_t)tap * CIN + K) * FOUT4 + N];
}

// ---------------------------------------------------------------- main step
// Block: 2*MG output rows x 16 cols x all 4F gate channels, 8 wave32.
// Wave (chunk, mg) owns the FOUR gate tiles of f-chunk `chunk` (nt = ga*CHUNKS+chunk)
// for rows 2*mg, 2*mg+1 -> all gates of a (pixel,channel) sit in one lane's registers,
// so the LSTM epilogue is pure in-register VALU (no z LDS round-trip, one barrier only).

template<int CIN, int F, bool LAST>
__global__ __launch_bounds__(256)
void convlstm_step(const _Float16* __restrict__ in_seq,  // [B,T,H,W,CIN] f16
                   const _Float16* __restrict__ hprev,   // [B,H,W,F] f16
                   _Float16*       __restrict__ hcur,    // [B,H,W,F] f16
                   float*          __restrict__ cstate,  // [B,H,W,F] f32
                   const _Float16* __restrict__ pwx,     // packed [9][CIN/32][NT][32][16]
                   const _Float16* __restrict__ pwh,     // packed [9][F/32][NT][32][16]
                   const float*    __restrict__ bias,    // [4F] f32
                   _Float16*       __restrict__ out_seq, // [B,T,H,W,F] relu(h) f16 (!LAST)
                   float*          __restrict__ out32,   // [B,T,H,W,F] relu(h) f32 (LAST)
                   int t) {
    constexpr int NT     = (4 * F) / 16;  // N tiles of 16
    constexpr int CHUNKS = F / 16;        // f-chunks (gate-channel groups of 16)
    constexpr int MG     = 8 / CHUNKS;    // row-groups (waves per chunk column)
    constexpr int ROWS   = 2 * MG;        // output rows per block
    constexpr int PR     = ROWS + 2;      // halo rows per block
    constexpr int KCX    = CIN / 32;      // K chunks for x-conv
    constexpr int KCH    = F / 32;        // K chunks for h-conv

    __shared__ _Float16 sx[PR * 18 * CIN];  // halo patch of x_t
    __shared__ _Float16 sh[PR * 18 * F];    // halo patch of h_{t-1}

    const int tid = threadIdx.x;
    const int x0 = blockIdx.x * 16, yb = blockIdx.y * ROWS, b = blockIdx.z;

    const _Float16* in = in_seq + ((size_t)(b * Tn + t) * Hn * Wn) * CIN;
    const _Float16* hp = hprev + (size_t)b * Hn * Wn * F;

    // cooperative zero-padded halo loads (b64 granularity, channel-contiguous)
    for (int e = tid; e < PR * 18 * CIN / 4; e += 256) {
        int r  = e / (18 * CIN / 4);
        int rm = e - r * (18 * CIN / 4);
        int c  = rm / (CIN / 4);
        int c4 = rm - c * (CIN / 4);
        int gy = yb + r - 1, gx = x0 + c - 1;
        uint2 v = {0u, 0u};
        if (gy >= 0 && gy < Hn && gx >= 0 && gx < Wn)
            v = *(const uint2*)(in + ((size_t)gy * Wn + gx) * CIN + c4 * 4);
        *(uint2*)(sx + e * 4) = v;
    }
    for (int e = tid; e < PR * 18 * F / 4; e += 256) {
        int r  = e / (18 * F / 4);
        int rm = e - r * (18 * F / 4);
        int c  = rm / (F / 4);
        int c4 = rm - c * (F / 4);
        int gy = yb + r - 1, gx = x0 + c - 1;
        uint2 v = {0u, 0u};
        if (gy >= 0 && gy < Hn && gx >= 0 && gx < Wn)
            v = *(const uint2*)(hp + ((size_t)gy * Wn + gx) * F + c4 * 4);
        *(uint2*)(sh + e * 4) = v;
    }
    __syncthreads();

    const int wave = tid >> 5, lane = tid & 31;
    const int m = lane & 15, hiq = lane >> 4;  // column index, half-wave select
    const int cbx = hiq * 8;                   // per-lane channel sub-offset
    const int chunk = wave % CHUNKS;           // which 16-channel f-chunk
    const int mg    = wave / CHUNKS;           // which pair of rows
    const int r0    = 2 * mg;                  // first block-relative output row

    // accumulators: 4 gates x 2 row-subtiles, seeded with per-gate bias
    v8f acc[4][2];
#pragma unroll
    for (int ga = 0; ga < 4; ++ga) {
        float bv = bias[ga * F + chunk * 16 + m];
#pragma unroll
        for (int s = 0; s < 2; ++s)
#pragma unroll
            for (int r = 0; r < 8; ++r) acc[ga][s][r] = bv;
    }

#pragma unroll
    for (int tap = 0; tap < 9; ++tap) {
        const int dy = tap / 3, dx = tap % 3;
        if (tap < 8) {  // warm L2/WGP$ for the next tap's weight fragments
            __builtin_prefetch(pwx + (((size_t)((tap + 1) * KCX) * NT + chunk) * 32 + lane) * 16, 0, 1);
            __builtin_prefetch(pwh + (((size_t)((tap + 1) * KCH) * NT + chunk) * 32 + lane) * 16, 0, 1);
        }
        // x-conv: A loaded per row-subtile (reused by 4 gate tiles), B reused by 2 rows
#pragma unroll
        for (int kc = 0; kc < KCX; ++kc) {
            v16h a0 = load_a(sx, r0 + 0 + dy, m + dx, CIN, kc * 32 + cbx);
            v16h a1 = load_a(sx, r0 + 1 + dy, m + dx, CIN, kc * 32 + cbx);
#pragma unroll
            for (int ga = 0; ga < 4; ++ga) {
                int nt = ga * CHUNKS + chunk;
                const v16h bf = *(const v16h*)(pwx + (((size_t)(tap * KCX + kc) * NT + nt) * 32 + lane) * 16);
                acc[ga][0] = __builtin_amdgcn_wmma_f32_16x16x32_f16(false, a0, false, bf,
                                                                    (short)0, acc[ga][0], false, false);
                acc[ga][1] = __builtin_amdgcn_wmma_f32_16x16x32_f16(false, a1, false, bf,
                                                                    (short)0, acc[ga][1], false, false);
            }
        }
        // h-conv
#pragma unroll
        for (int kc = 0; kc < KCH; ++kc) {
            v16h a0 = load_a(sh, r0 + 0 + dy, m + dx, F, kc * 32 + cbx);
            v16h a1 = load_a(sh, r0 + 1 + dy, m + dx, F, kc * 32 + cbx);
#pragma unroll
            for (int ga = 0; ga < 4; ++ga) {
                int nt = ga * CHUNKS + chunk;
                const v16h bf = *(const v16h*)(pwh + (((size_t)(tap * KCH + kc) * NT + nt) * 32 + lane) * 16);
                acc[ga][0] = __builtin_amdgcn_wmma_f32_16x16x32_f16(false, a0, false, bf,
                                                                    (short)0, acc[ga][0], false, false);
                acc[ga][1] = __builtin_amdgcn_wmma_f32_16x16x32_f16(false, a1, false, bf,
                                                                    (short)0, acc[ga][1], false, false);
            }
        }
    }

    // in-register LSTM epilogue: C/D layout element (col = r + 8*hiq, ch = chunk*16 + m)
    const int chv = chunk * 16 + m;
    float*    cp = cstate + (size_t)b * Hn * Wn * F;
    _Float16* hc = hcur + (size_t)b * Hn * Wn * F;
#pragma unroll
    for (int s = 0; s < 2; ++s) {
        const int row = yb + r0 + s;
#pragma unroll
        for (int r = 0; r < 8; ++r) {
            const int px = x0 + r + 8 * hiq;
            size_t pix = ((size_t)row * Wn + px) * F + chv;
            float ig = hsig(acc[0][s][r]);
            float fg = hsig(acc[1][s][r]);
            float gg = tanhf(acc[2][s][r]);
            float og = hsig(acc[3][s][r]);
            float cn = fg * cp[pix] + ig * gg;
            float hn = og * tanhf(cn);
            cp[pix] = cn;
            hc[pix] = (_Float16)hn;
            float rl = fmaxf(hn, 0.0f);
            size_t oidx = (size_t)(b * Tn + t) * Hn * Wn * F + pix;
            if (LAST) out32[oidx] = rl;
            else      out_seq[oidx] = (_Float16)rl;
        }
    }
}

// ---------------------------------------------------------------- host side

extern "C" void kernel_launch(void* const* d_in, const int* in_sizes, int n_in,
                              void* d_out, int out_size, void* d_ws, size_t ws_size,
                              hipStream_t stream) {
    (void)in_sizes; (void)n_in; (void)out_size; (void)ws_size;

    const float* x   = (const float*)d_in[0];
    const float* Wx1 = (const float*)d_in[1];
    const float* Wh1 = (const float*)d_in[2];
    const float* b1  = (const float*)d_in[3];
    const float* Wx2 = (const float*)d_in[4];
    const float* Wh2 = (const float*)d_in[5];
    const float* b2  = (const float*)d_in[6];
    const float* Wx3 = (const float*)d_in[7];
    const float* Wh3 = (const float*)d_in[8];
    const float* b3  = (const float*)d_in[9];

    char* ws = (char*)d_ws;
    size_t off = 0;
    auto alloc = [&](size_t bytes) -> void* {
        void* p = ws + off;
        off = (off + bytes + 255) & ~(size_t)255;
        return p;
    };

    const size_t PIX = (size_t)Bn * Hn * Wn;          // 16384 pixels
    const size_t SEQ = (size_t)Bn * Tn * Hn * Wn;     // 262144 pixel-steps

    _Float16* xf16 = (_Float16*)alloc(SEQ * 64 * 2);  // x as f16
    _Float16* seq1 = (_Float16*)alloc(SEQ * 64 * 2);  // relu(h) layer-1
    _Float16* seq2 = (_Float16*)alloc(SEQ * 32 * 2);  // relu(h) layer-2
    _Float16* hA   = (_Float16*)alloc(PIX * 64 * 2);  // h ping
    _Float16* hB   = (_Float16*)alloc(PIX * 64 * 2);  // h pong
    float*    cst  = (float*)   alloc(PIX * 64 * 4);  // c state
    _Float16* pwx1 = (_Float16*)alloc((size_t)9 * 64 * 256 * 2);
    _Float16* pwh1 = (_Float16*)alloc((size_t)9 * 64 * 256 * 2);
    _Float16* pwx2 = (_Float16*)alloc((size_t)9 * 64 * 128 * 2);
    _Float16* pwh2 = (_Float16*)alloc((size_t)9 * 32 * 128 * 2);
    _Float16* pwx3 = (_Float16*)alloc((size_t)9 * 32 * 256 * 2);
    _Float16* pwh3 = (_Float16*)alloc((size_t)9 * 64 * 256 * 2);

    // weight packing + input conversion
    {
        int n;
        n = 9 * 64 * 256; pack_w<64, 256><<<(n + 255) / 256, 256, 0, stream>>>(Wx1, pwx1);
        n = 9 * 64 * 256; pack_w<64, 256><<<(n + 255) / 256, 256, 0, stream>>>(Wh1, pwh1);
        n = 9 * 64 * 128; pack_w<64, 128><<<(n + 255) / 256, 256, 0, stream>>>(Wx2, pwx2);
        n = 9 * 32 * 128; pack_w<32, 128><<<(n + 255) / 256, 256, 0, stream>>>(Wh2, pwh2);
        n = 9 * 32 * 256; pack_w<32, 256><<<(n + 255) / 256, 256, 0, stream>>>(Wx3, pwx3);
        n = 9 * 64 * 256; pack_w<64, 256><<<(n + 255) / 256, 256, 0, stream>>>(Wh3, pwh3);
        n = (int)(SEQ * 64);
        cvt_f32_to_f16<<<(n + 255) / 256, 256, 0, stream>>>(x, xf16, n);
    }

    dim3 blk(256);
    dim3 grid1(Wn / 16, Hn / 4, Bn);  // F=64: 4 rows/block
    dim3 grid2(Wn / 16, Hn / 8, Bn);  // F=32: 8 rows/block
    dim3 grid3(Wn / 16, Hn / 4, Bn);  // F=64: 4 rows/block

    // ---- layer 1: CIN=64, F=64
    hipMemsetAsync(hA, 0, PIX * 64 * 2, stream);
    hipMemsetAsync(cst, 0, PIX * 64 * 4, stream);
    for (int t = 0; t < Tn; ++t) {
        const _Float16* hp = (t & 1) ? hB : hA;
        _Float16*       hc = (t & 1) ? hA : hB;
        convlstm_step<64, 64, false><<<grid1, blk, 0, stream>>>(
            xf16, hp, hc, cst, pwx1, pwh1, b1, seq1, nullptr, t);
    }

    // ---- layer 2: CIN=64, F=32
    hipMemsetAsync(hA, 0, PIX * 32 * 2, stream);
    hipMemsetAsync(cst, 0, PIX * 32 * 4, stream);
    for (int t = 0; t < Tn; ++t) {
        const _Float16* hp = (t & 1) ? hB : hA;
        _Float16*       hc = (t & 1) ? hA : hB;
        convlstm_step<64, 32, false><<<grid2, blk, 0, stream>>>(
            seq1, hp, hc, cst, pwx2, pwh2, b2, seq2, nullptr, t);
    }

    // ---- layer 3: CIN=32, F=64 -> f32 output
    hipMemsetAsync(hA, 0, PIX * 64 * 2, stream);
    hipMemsetAsync(cst, 0, PIX * 64 * 4, stream);
    for (int t = 0; t < Tn; ++t) {
        const _Float16* hp = (t & 1) ? hB : hA;
        _Float16*       hc = (t & 1) ? hA : hB;
        convlstm_step<32, 64, true><<<grid3, blk, 0, stream>>>(
            seq2, hp, hc, cst, pwx3, pwh3, b3, nullptr, (float*)d_out, t);
    }
}
